// GPTJAttention_55327768708099
// MI455X (gfx1250) — matrix operand order repaired and checked by reading, not compiled, not run on hardware
//
#include <hip/hip_runtime.h>

typedef _Float16 f16;
typedef _Float16 v16h __attribute__((ext_vector_type(16)));
typedef _Float16 v8h  __attribute__((ext_vector_type(8)));
typedef _Float16 v4h  __attribute__((ext_vector_type(4)));
typedef float    v8f  __attribute__((ext_vector_type(8)));

#define EMBED  4096
#define NHEADS 16
#define HDIM   256
#define ROT    64
#define BATCH  2
#define SEQ    2048

union AF { v16h v; v8h h[2]; };

static __device__ inline v8f wmma_f16(v16h a, v16h b, v8f c) {
  return __builtin_amdgcn_wmma_f32_16x16x32_f16(false, a, false, b, (short)0, c, false, false);
}

// ---------------------------------------------------------------- f32 -> f16
__global__ void cvt_f32_f16_k(const float* __restrict__ in, f16* __restrict__ out, int n4) {
  int i = blockIdx.x * blockDim.x + threadIdx.x;
  if (i < n4) {
    float4 v = ((const float4*)in)[i];
    v4h o;
    o[0] = (f16)v.x; o[1] = (f16)v.y; o[2] = (f16)v.z; o[3] = (f16)v.w;
    ((v4h*)out)[i] = o;
  }
}

// ------------------------------------------- C[M,N] = A[M,K] * B[N,K]^T (f32 out)
// Async-LDS double-buffered pipeline. Block = 256 thr (8 waves), 128x128 tile.
// Per K-step(32): block async-copies A/B 128x32 tiles into LDS (16 KB unique data,
// moved once), waves read fragments via ds_load_b128, 8 WMMAs per wave.
#define ROWPAD 40   /* halves per LDS row: 32 data + 8 pad (16B) to spread banks */

__global__ __launch_bounds__(256)
void wmma_gemm_nt(const f16* __restrict__ A, const f16* __restrict__ Bw,
                  float* __restrict__ C, int M, int N, int K) {
  __shared__ __attribute__((aligned(16))) f16 sA[2][128 * ROWPAD];
  __shared__ __attribute__((aligned(16))) f16 sB[2][128 * ROWPAD];

  const int tid  = threadIdx.x;
  const int lane = tid & 31;
  const int wid  = tid >> 5;
  const int l16  = lane & 15;
  const int half = lane >> 4;
  const int bm = blockIdx.y * 128;
  const int bn = blockIdx.x * 128;
  const int wm = (wid & 3) * 32;
  const int wn = (wid >> 2) * 64;

  // copy assignment: chunk = 16B; 4 chunks per 64B row; 2 chunks per thread per tile
  const int c0   = tid * 2;
  const int row0 = c0 >> 2,       col0 = (c0 & 3);
  const int row1 = (c0 + 1) >> 2, col1 = ((c0 + 1) & 3);

  const f16* Ab = A  + (size_t)bm * K;
  const f16* Bb = Bw + (size_t)bn * K;
  const unsigned sA0 = (unsigned)(size_t)&sA[0][0];
  const unsigned sB0 = (unsigned)(size_t)&sB[0][0];
  const unsigned bufStride = 128u * ROWPAD * sizeof(f16);   // 10240 B

  v8f acc[2][4];
  #pragma unroll
  for (int i = 0; i < 2; ++i)
    #pragma unroll
    for (int j = 0; j < 4; ++j) acc[i][j] = (v8f){0,0,0,0,0,0,0,0};

  auto issue = [&](int kt, int p) {
    const int k0 = kt * 32;
    const f16* ga0 = Ab + (size_t)row0 * K + k0 + col0 * 8;
    const f16* ga1 = Ab + (size_t)row1 * K + k0 + col1 * 8;
    const f16* gb0 = Bb + (size_t)row0 * K + k0 + col0 * 8;
    const f16* gb1 = Bb + (size_t)row1 * K + k0 + col1 * 8;
    const unsigned la0 = sA0 + (unsigned)p * bufStride + row0 * 80 + col0 * 16;
    const unsigned la1 = sA0 + (unsigned)p * bufStride + row1 * 80 + col1 * 16;
    const unsigned lb0 = sB0 + (unsigned)p * bufStride + row0 * 80 + col0 * 16;
    const unsigned lb1 = sB0 + (unsigned)p * bufStride + row1 * 80 + col1 * 16;
    asm volatile("global_load_async_to_lds_b128 %0, %1, off" :: "v"(la0), "v"(ga0) : "memory");
    asm volatile("global_load_async_to_lds_b128 %0, %1, off" :: "v"(la1), "v"(ga1) : "memory");
    asm volatile("global_load_async_to_lds_b128 %0, %1, off" :: "v"(lb0), "v"(gb0) : "memory");
    asm volatile("global_load_async_to_lds_b128 %0, %1, off" :: "v"(lb1), "v"(gb1) : "memory");
  };

  const int NT = K / 32;
  issue(0, 0);
  for (int kt = 0; kt < NT; ++kt) {
    const int p = kt & 1;
    asm volatile("s_wait_asynccnt 0x0" ::: "memory");  // my stage-kt copies landed
    __syncthreads();                                    // everyone's copies landed;
                                                        // buffer p^1 readers (kt-1) done
    if (kt + 1 < NT) issue(kt + 1, p ^ 1);              // overlap next fill with compute

    const f16* aP = &sA[p][0];
    const f16* bP = &sB[p][0];
    AF a[2], b[4];
    #pragma unroll
    for (int i = 0; i < 2; ++i) {
      const f16* ar = aP + (wm + i * 16 + l16) * ROWPAD + half * 16;
      a[i].h[0] = *(const v8h*)ar;
      a[i].h[1] = *(const v8h*)(ar + 8);
    }
    #pragma unroll
    for (int j = 0; j < 4; ++j)
      b[j].v = *(const v16h*)(bP + (wn + j * 16 + l16) * ROWPAD + half * 16);
    #pragma unroll
    for (int i = 0; i < 2; ++i)
      #pragma unroll
      for (int j = 0; j < 4; ++j)
        acc[i][j] = wmma_f16(a[i].v, b[j].v, acc[i][j]);
  }

  // C-layout: elem r -> row bm+wm+i*16 + r + 8*half, col bn+wn+j*16 + l16
  #pragma unroll
  for (int i = 0; i < 2; ++i)
    #pragma unroll
    for (int j = 0; j < 4; ++j) {
      const int rbase = bm + wm + i * 16 + 8 * half;
      const int col   = bn + wn + j * 16 + l16;
      #pragma unroll
      for (int r = 0; r < 8; ++r)
        C[(size_t)(rbase + r) * N + col] = acc[i][j][r];
    }
}

// --------------------- interleaved GPT-J RoPE + repack [B,S,H,D]f32 -> [B,H,S,D]f16
__global__ void rope_pack_k(const float* __restrict__ src, const int* __restrict__ pos,
                            const float* __restrict__ embed, f16* __restrict__ dst) {
  int idx = blockIdx.x * blockDim.x + threadIdx.x;   // B*S*H*(D/2) = 8,388,608
  const int PAIRS = HDIM / 2;
  int i = idx & (PAIRS - 1);
  int t = idx / PAIRS;
  int h = t & (NHEADS - 1); t >>= 4;
  int s = t & (SEQ - 1);
  int b = t >> 11;
  const float* row = src + (((size_t)(b * SEQ + s)) * NHEADS + h) * HDIM;
  float x0 = row[2 * i], x1 = row[2 * i + 1];
  float o0 = x0, o1 = x1;
  if (i < ROT / 2) {
    int p = pos[b * SEQ + s];
    float sn = embed[p * ROT + i];
    float cs = embed[p * ROT + ROT / 2 + i];
    o0 = x0 * cs - x1 * sn;
    o1 = x1 * cs + x0 * sn;
  }
  f16* orow = dst + (((size_t)(b * NHEADS + h)) * SEQ + s) * HDIM;
  orow[2 * i]     = (f16)o0;
  orow[2 * i + 1] = (f16)o1;
}

// --------------------- V repack: [B,S,H,D] f32 -> [B,H,D,S] f16 (transposed)
__global__ void vpack_t_k(const float* __restrict__ src, f16* __restrict__ dst) {
  size_t idx = (size_t)blockIdx.x * blockDim.x + threadIdx.x;  // B*H*D*S
  int s = (int)(idx & (SEQ - 1));
  size_t t = idx >> 11;
  int d = (int)(t & (HDIM - 1)); t >>= 8;
  int h = (int)(t & (NHEADS - 1));
  int b = (int)(t >> 4);
  dst[idx] = (f16)src[(((size_t)(b * SEQ + s)) * NHEADS + h) * HDIM + d];
}

// --------------------- flash attention, wave = 16 queries, scoresT = K * Q^T
__global__ __launch_bounds__(256)
void flash_attn_k(const f16* __restrict__ Q,   // [B,H,S,D]
                  const f16* __restrict__ K,   // [B,H,S,D]
                  const f16* __restrict__ Vt,  // [B,H,D,S]
                  f16* __restrict__ O) {       // [B,S,E]
  const int lane = threadIdx.x & 31;
  const int wid  = threadIdx.x >> 5;
  const int l16  = lane & 15;
  const int half = lane >> 4;
  const int qc = blockIdx.x & 15;   // S/128 chunks
  const int bh = blockIdx.x >> 4;
  const int b  = bh >> 4;
  const int h  = bh & 15;
  const int q0 = qc * 128 + wid * 16;

  const f16* Qb = Q  + (size_t)bh * SEQ * HDIM;
  const f16* Kb = K  + (size_t)bh * SEQ * HDIM;
  const f16* Vb = Vt + (size_t)bh * HDIM * SEQ;

  // Q as B-fragments (Bfrag[d][q] = Q[q][d]): lane col q = l16, 16 contiguous d
  v16h qf[8];
  #pragma unroll
  for (int dt = 0; dt < 8; ++dt)
    qf[dt] = *(const v16h*)(Qb + (size_t)(q0 + l16) * HDIM + dt * 32 + half * 16);

  v8f o[16];
  #pragma unroll
  for (int dt = 0; dt < 16; ++dt) o[dt] = (v8f){0,0,0,0,0,0,0,0};
  float m = -3.0e38f, l = 0.0f;

  for (int kb = 0; kb < q0 + 16; kb += 32) {
    v8f pt0 = (v8f){0,0,0,0,0,0,0,0};
    v8f pt1 = (v8f){0,0,0,0,0,0,0,0};
    #pragma unroll
    for (int dt = 0; dt < 8; ++dt) {
      AF ka0, ka1;
      const f16* kp = Kb + (size_t)(kb + l16) * HDIM + dt * 32 + half * 16;
      ka0.h[0] = *(const v8h*)kp;
      ka0.h[1] = *(const v8h*)(kp + 8);
      const f16* kp1 = kp + (size_t)16 * HDIM;
      ka1.h[0] = *(const v8h*)kp1;
      ka1.h[1] = *(const v8h*)(kp1 + 8);
      pt0 = wmma_f16(ka0.v, qf[dt], pt0);   // scoresT[key 0..15][q]
      pt1 = wmma_f16(ka1.v, qf[dt], pt1);   // scoresT[key 16..31][q]
    }
    const int q = q0 + l16;
    float tmax = -3.0e38f;
    #pragma unroll
    for (int r = 0; r < 8; ++r) {
      int k0r = kb + r + 8 * half;
      float s0 = pt0[r] * 0.0625f;          // 1/sqrt(256)
      if (k0r > q) s0 = -3.0e38f;
      float s1 = pt1[r] * 0.0625f;
      if (k0r + 16 > q) s1 = -3.0e38f;
      pt0[r] = s0; pt1[r] = s1;
      tmax = fmaxf(tmax, fmaxf(s0, s1));
    }
    tmax = fmaxf(tmax, __shfl_xor(tmax, 16, 32));
    float mnew = fmaxf(m, tmax);
    float corr = __expf(m - mnew);
    float rsum = 0.0f;
    AF pa;                                   // P as A-fragment: pure per-lane repack
    #pragma unroll
    for (int r = 0; r < 8; ++r) {
      float p0 = __expf(pt0[r] - mnew);
      float p1 = __expf(pt1[r] - mnew);
      rsum += p0 + p1;
      pa.v[r]     = (f16)p0;                 // kk = half*8 + r
      pa.v[8 + r] = (f16)p1;                 // kk = 16 + half*8 + r
    }
    rsum += __shfl_xor(rsum, 16, 32);
    l = l * corr + rsum;
    m = mnew;
    float osc[8];
    #pragma unroll
    for (int r = 0; r < 8; ++r)              // O vgpr r belongs to query r + 8*half
      osc[r] = __shfl(corr, r + 8 * half, 32);
    #pragma unroll
    for (int dt = 0; dt < 16; ++dt) {
      AF vb;
      const f16* vp = Vb + (size_t)(dt * 16 + l16) * SEQ + kb + half * 8;
      vb.h[0] = *(const v8h*)vp;
      vb.h[1] = *(const v8h*)(vp + 16);
      #pragma unroll
      for (int r = 0; r < 8; ++r) o[dt][r] *= osc[r];
      o[dt] = wmma_f16(pa.v, vb.v, o[dt]);
    }
  }

  float invl = 1.0f / l;
  float winv[8];
  #pragma unroll
  for (int r = 0; r < 8; ++r) winv[r] = __shfl(invl, r + 8 * half, 32);
  #pragma unroll
  for (int dt = 0; dt < 16; ++dt)
    #pragma unroll
    for (int r = 0; r < 8; ++r) {
      int srow = q0 + r + 8 * half;
      int col  = h * HDIM + dt * 16 + l16;
      O[((size_t)(b * SEQ + srow)) * EMBED + col] = (f16)(o[dt][r] * winv[r]);
    }
}

// ----------------------------------------------------------------------------
extern "C" void kernel_launch(void* const* d_in, const int* in_sizes, int n_in,
                              void* d_out, int out_size, void* d_ws, size_t ws_size,
                              hipStream_t stream) {
  (void)in_sizes; (void)n_in; (void)out_size; (void)ws_size;
  const float* hidden = (const float*)d_in[0];
  const int*   pos    = (const int*)d_in[1];
  const float* qw     = (const float*)d_in[2];
  const float* kw     = (const float*)d_in[3];
  const float* vw     = (const float*)d_in[4];
  const float* ow     = (const float*)d_in[5];
  const float* embed  = (const float*)d_in[6];

  const size_t BSE = (size_t)BATCH * SEQ * EMBED;   // 16,777,216
  const size_t E2  = (size_t)EMBED * EMBED;

  char* ws = (char*)d_ws;
  size_t off = 0;
  auto take = [&](size_t bytes) { char* p = ws + off; off += (bytes + 255) & ~(size_t)255; return p; };
  f16*   h16    = (f16*)take(BSE * 2);
  f16*   qw16   = (f16*)take(E2 * 2);
  f16*   kw16   = (f16*)take(E2 * 2);
  f16*   vw16   = (f16*)take(E2 * 2);
  f16*   ow16   = (f16*)take(E2 * 2);
  float* qf32   = (float*)take(BSE * 4);
  float* kf32   = (float*)take(BSE * 4);
  float* vf32   = (float*)take(BSE * 4);
  f16*   q16    = (f16*)take(BSE * 2);    // [B,H,S,D]
  f16*   k16    = (f16*)take(BSE * 2);    // [B,H,S,D]
  f16*   vt16   = (f16*)take(BSE * 2);    // [B,H,D,S]
  f16*   attn16 = (f16*)take(BSE * 2);    // [B,S,E]

  const int n4 = (int)(BSE / 4);          // == E2/4 too
  const int cvtBlocks = n4 / 256;
  cvt_f32_f16_k<<<cvtBlocks, 256, 0, stream>>>(hidden, h16, n4);
  cvt_f32_f16_k<<<cvtBlocks, 256, 0, stream>>>(qw, qw16, n4);
  cvt_f32_f16_k<<<cvtBlocks, 256, 0, stream>>>(kw, kw16, n4);
  cvt_f32_f16_k<<<cvtBlocks, 256, 0, stream>>>(vw, vw16, n4);
  cvt_f32_f16_k<<<cvtBlocks, 256, 0, stream>>>(ow, ow16, n4);

  const int Mrows = BATCH * SEQ;          // 4096
  dim3 ggrid(EMBED / 128, Mrows / 128);   // 32 x 32
  wmma_gemm_nt<<<ggrid, 256, 0, stream>>>(h16, qw16, qf32, Mrows, EMBED, EMBED);
  wmma_gemm_nt<<<ggrid, 256, 0, stream>>>(h16, kw16, kf32, Mrows, EMBED, EMBED);
  wmma_gemm_nt<<<ggrid, 256, 0, stream>>>(h16, vw16, vf32, Mrows, EMBED, EMBED);

  const int ropeBlocks = (BATCH * SEQ * NHEADS * (HDIM / 2)) / 256;  // 32768
  rope_pack_k<<<ropeBlocks, 256, 0, stream>>>(qf32, pos, embed, q16);
  rope_pack_k<<<ropeBlocks, 256, 0, stream>>>(kf32, pos, embed, k16);
  vpack_t_k<<<(int)(BSE / 256), 256, 0, stream>>>(vf32, vt16);

  const int attnBlocks = BATCH * NHEADS * (SEQ / 128);               // 512
  flash_attn_k<<<attnBlocks, 256, 0, stream>>>(q16, k16, vt16, attn16);

  wmma_gemm_nt<<<ggrid, 256, 0, stream>>>(attn16, ow16, (float*)d_out, Mrows, EMBED, EMBED);
}
